// MinkowskiCE_53790170415572
// MI455X (gfx1250) — compile-verified, matrix operand order checked by reading, not run on hardware
//
#include <hip/hip_runtime.h>
#include <hip/hip_bf16.h>

// ---------------------------------------------------------------------------
// Types for CDNA5 WMMA (wave32): D(16x16 f32) = A(16x32 bf16) x B(32x16 bf16) + C
// ---------------------------------------------------------------------------
typedef __bf16 bf16_t;
typedef __bf16 v16bf __attribute__((ext_vector_type(16)));
typedef __bf16 v2bf  __attribute__((ext_vector_type(2)));
typedef float  v8f   __attribute__((ext_vector_type(8)));

__device__ __forceinline__ bf16_t f2bf(float f) {
    // round-to-nearest-even f32 -> bf16, bit-exact and guaranteed to compile
    union { float f; unsigned u; } a; a.f = f;
    unsigned u = a.u;
    u += 0x7FFFu + ((u >> 16) & 1u);
    union { unsigned short s; bf16_t b; } o; o.s = (unsigned short)(u >> 16);
    return o.b;
}

__device__ __forceinline__ unsigned pack2bf(float a, float b) {
    union { v2bf v; unsigned u; } o;
    o.v[0] = f2bf(a);
    o.v[1] = f2bf(b);
    return o.u;
}

#define NEG_SLOPE 0.01f
#define KOFF 27

// ---------------------------------------------------------------------------
// Sparse conv via gather + WMMA.
//   out[j, :] = sum_k mask[k,j] ? x[nbr[k,j], :] @ W[k] : 0
// x: [Nin, Cin] f32, nbr/mask: [27, Nout], W: [27, Cin, Cout], out: [Nout, Cout]
// Block = 128 threads (4 waves). Block owns 32 output rows x 128 cols.
// Wave w owns a 32x32 tile: cols [(blockIdx.y*4 + w)*32, +32).
// 2x2 register tiling: 2 A fragments (rows 0-15 / 16-31, from LDS) x
// 2 B fragments (two 16-col spans, from L2-resident weights) -> 4 WMMAs
// per staged tile, doubling FLOPs per fetched byte vs a 1x2 tiling.
// A tile is staged in LDS as packed bf16 pairs (b32 words).
// ---------------------------------------------------------------------------
__global__ __launch_bounds__(128)
void sparse_conv_wmma_kernel(const float* __restrict__ x,
                             const int* __restrict__ nbr,
                             const unsigned char* __restrict__ mask,
                             const float* __restrict__ W,
                             float* __restrict__ out,
                             int Nin, int Nout, int Cin, int Cout)
{
    (void)Nin;
    __shared__ unsigned Atile[32][16];   // 32 rows x 32 bf16 chans (packed pairs)

    const int tid  = threadIdx.x;
    const int wave = tid >> 5;
    const int lane = tid & 31;
    const int half = lane >> 4;          // 0: lanes 0-15, 1: lanes 16-31
    const int row0 = blockIdx.x * 32;
    const int col0 = (blockIdx.y * 4 + wave) * 32;
    const bool active0 = (col0 < Cout);        // first 16-col span
    const bool active1 = (col0 + 16 < Cout);   // second 16-col span
    const int colA = col0 + (lane & 15);
    const int colB = colA + 16;
    const int kchunks = (Cin + 31) >> 5;

    // Staging role: thread t gathers row (t>>2), channel octet (t&3) of chunk.
    const int srow = tid >> 2;           // 0..31
    const int squad = tid & 3;           // 0..3

    v8f acc00 = {}, acc01 = {}, acc10 = {}, acc11 = {};

    for (int k = 0; k < KOFF; ++k) {
        const int*           nbr_k = nbr  + (size_t)k * Nout;
        const unsigned char* msk_k = mask + (size_t)k * Nout;
        const float*         W_k   = W    + (size_t)k * Cin * Cout;
        if (k + 1 < KOFF)  // hint next weight slab toward caches
            __builtin_prefetch(W + (size_t)(k + 1) * Cin * Cout, 0, 0);

        // Per-k gather metadata for this thread's staging row (hoisted out of
        // the Cin-chunk loop).
        bool valid = false; int src = 0;
        {
            const int row = row0 + srow;
            if (row < Nout && msk_k[row]) { valid = true; src = nbr_k[row]; }
        }
        const float* xrow = x + (size_t)src * Cin;

        for (int kc = 0; kc < kchunks; ++kc) {
            const int ch0 = kc << 5;
            const int chb = ch0 + squad * 8;
            __syncthreads();
            // Stage gathered A tile: 32 rows x 32 chans, masked & zero-padded.
            {
                float v[8];
                if (valid && (ch0 + 32) <= Cin) {
                    const float4 f0 = *(const float4*)(xrow + chb);
                    const float4 f1 = *(const float4*)(xrow + chb + 4);
                    v[0] = f0.x; v[1] = f0.y; v[2] = f0.z; v[3] = f0.w;
                    v[4] = f1.x; v[5] = f1.y; v[6] = f1.z; v[7] = f1.w;
                } else {
#pragma unroll
                    for (int j = 0; j < 8; ++j) {
                        const int ch = chb + j;
                        v[j] = (valid && ch < Cin) ? xrow[ch] : 0.f;
                    }
                }
#pragma unroll
                for (int j = 0; j < 4; ++j)
                    Atile[srow][squad * 4 + j] = pack2bf(v[2 * j], v[2 * j + 1]);
            }
            __syncthreads();
            if (!active0) continue;

            // Two A fragments (rows 0-15, rows 16-31) from packed LDS words.
            // 16-bit A layout: elems 0..7 -> K = half*8 + i,
            //                  elems 8..15 -> K = 16 + half*8 + i
            v16bf a0, a1;
            {
                const int r = lane & 15;
#pragma unroll
                for (int j = 0; j < 4; ++j) {
                    union { unsigned u; v2bf v; } lo0, hi0, lo1, hi1;
                    lo0.u = Atile[r][half * 4 + j];
                    hi0.u = Atile[r][8 + half * 4 + j];
                    lo1.u = Atile[r + 16][half * 4 + j];
                    hi1.u = Atile[r + 16][8 + half * 4 + j];
                    a0[2 * j]         = lo0.v[0];
                    a0[2 * j + 1]     = lo0.v[1];
                    a0[8 + 2 * j]     = hi0.v[0];
                    a0[8 + 2 * j + 1] = hi0.v[1];
                    a1[2 * j]         = lo1.v[0];
                    a1[2 * j + 1]     = lo1.v[1];
                    a1[8 + 2 * j]     = hi1.v[0];
                    a1[8 + 2 * j + 1] = hi1.v[1];
                }
            }
            // B fragment, first 16-col span (32x16): lane = col (mod 16),
            // same K striping as A.
            const int kb = half * 8;
            v16bf b0;
            {
#pragma unroll
                for (int i = 0; i < 8; ++i) {
                    const int kk0 = ch0 + kb + i;
                    const int kk1 = ch0 + 16 + kb + i;
                    const float w0 = (kk0 < Cin) ? W_k[(size_t)kk0 * Cout + colA] : 0.f;
                    const float w1 = (kk1 < Cin) ? W_k[(size_t)kk1 * Cout + colA] : 0.f;
                    b0[i]     = f2bf(w0);
                    b0[8 + i] = f2bf(w1);
                }
            }
            acc00 = __builtin_amdgcn_wmma_f32_16x16x32_bf16(
                false, a0, false, b0, (short)0, acc00, false, false);
            acc10 = __builtin_amdgcn_wmma_f32_16x16x32_bf16(
                false, a1, false, b0, (short)0, acc10, false, false);

            if (active1) {  // wave-uniform: EXEC stays all-ones around WMMA
                v16bf b1;
#pragma unroll
                for (int i = 0; i < 8; ++i) {
                    const int kk0 = ch0 + kb + i;
                    const int kk1 = ch0 + 16 + kb + i;
                    const float w0 = (kk0 < Cin) ? W_k[(size_t)kk0 * Cout + colB] : 0.f;
                    const float w1 = (kk1 < Cin) ? W_k[(size_t)kk1 * Cout + colB] : 0.f;
                    b1[i]     = f2bf(w0);
                    b1[8 + i] = f2bf(w1);
                }
                acc01 = __builtin_amdgcn_wmma_f32_16x16x32_bf16(
                    false, a0, false, b1, (short)0, acc01, false, false);
                acc11 = __builtin_amdgcn_wmma_f32_16x16x32_bf16(
                    false, a1, false, b1, (short)0, acc11, false, false);
            }
        }
    }

    if (active0) {
        // D layout: lanes 0-15 -> M = v, lanes 16-31 -> M = v + 8, N = lane&15
#pragma unroll
        for (int vi = 0; vi < 8; ++vi) {
            const int m = vi + half * 8;
            const int r0 = row0 + m;
            const int r1 = row0 + 16 + m;
            if (r0 < Nout) {
                out[(size_t)r0 * Cout + colA] = acc00[vi];
                if (active1) out[(size_t)r0 * Cout + colB] = acc01[vi];
            }
            if (r1 < Nout) {
                out[(size_t)r1 * Cout + colA] = acc10[vi];
                if (active1) out[(size_t)r1 * Cout + colB] = acc11[vi];
            }
        }
    }
}

// ---------------------------------------------------------------------------
// Glue kernels
// ---------------------------------------------------------------------------
__global__ void zero_kernel(float* __restrict__ p, size_t n) {
    for (size_t i = blockIdx.x * (size_t)blockDim.x + threadIdx.x; i < n;
         i += (size_t)gridDim.x * blockDim.x) p[i] = 0.f;
}

__global__ void mlp1_kernel(const float* __restrict__ feats,
                            const float* __restrict__ W,
                            float* __restrict__ out, int N) {
    int n = blockIdx.x * blockDim.x + threadIdx.x;
    if (n >= N) return;
    const float f0 = feats[n * 3 + 0], f1 = feats[n * 3 + 1], f2 = feats[n * 3 + 2];
#pragma unroll
    for (int c = 0; c < 16; ++c)
        out[(size_t)n * 16 + c] = f0 * W[c] + f1 * W[16 + c] + f2 * W[32 + c];
}

// per-channel sum & sumsq; one block per channel, LDS tree reduce
__global__ void bn_stats_kernel(const float* __restrict__ x, int N, int C,
                                float* __restrict__ stats) {
    __shared__ float s_sum[256], s_sq[256];
    const int c = blockIdx.x;
    float sum = 0.f, sq = 0.f;
    for (int r = threadIdx.x; r < N; r += blockDim.x) {
        const float v = x[(size_t)r * C + c];
        sum += v; sq += v * v;
    }
    s_sum[threadIdx.x] = sum; s_sq[threadIdx.x] = sq;
    __syncthreads();
    for (int s = 128; s > 0; s >>= 1) {
        if (threadIdx.x < s) {
            s_sum[threadIdx.x] += s_sum[threadIdx.x + s];
            s_sq[threadIdx.x]  += s_sq[threadIdx.x + s];
        }
        __syncthreads();
    }
    if (threadIdx.x == 0) { stats[c] = s_sum[0]; stats[C + c] = s_sq[0]; }
}

__global__ void bn_lrelu_kernel(float* __restrict__ x, int N, int C,
                                const float* __restrict__ stats,
                                const float* __restrict__ g,
                                const float* __restrict__ b) {
    const size_t total = (size_t)N * C;
    const float invN = 1.f / (float)N;
    for (size_t i = blockIdx.x * (size_t)blockDim.x + threadIdx.x; i < total;
         i += (size_t)gridDim.x * blockDim.x) {
        const int c = (int)(i % C);
        const float mean = stats[c] * invN;
        float var = stats[C + c] * invN - mean * mean;
        var = fmaxf(var, 0.f);
        float v = (x[i] - mean) * rsqrtf(var + 1e-5f) * g[c] + b[c];
        x[i] = v > 0.f ? v : NEG_SLOPE * v;
    }
}

__global__ void seg_accum_kernel(const float* __restrict__ x,
                                 const int* __restrict__ sid, int N, int C,
                                 float* __restrict__ sums,
                                 float* __restrict__ counts, int do_count) {
    const size_t total = (size_t)N * C;
    for (size_t i = blockIdx.x * (size_t)blockDim.x + threadIdx.x; i < total;
         i += (size_t)gridDim.x * blockDim.x) {
        const int n = (int)(i / C), c = (int)(i % C);
        const int v = sid[n];
        atomicAdd(&sums[(size_t)v * C + c], x[i]);
        if (do_count && c == 0) atomicAdd(&counts[v], 1.f);
    }
}

__global__ void seg_div_kernel(float* __restrict__ sums,
                               const float* __restrict__ counts, int NV, int C) {
    const size_t total = (size_t)NV * C;
    for (size_t i = blockIdx.x * (size_t)blockDim.x + threadIdx.x; i < total;
         i += (size_t)gridDim.x * blockDim.x)
        sums[i] /= fmaxf(counts[i / C], 1.f);
}

__global__ void maxpool_kernel(const float* __restrict__ x,
                               const int* __restrict__ nbr,
                               const unsigned char* __restrict__ mask,
                               float* __restrict__ out, int Nout, int C) {
    const size_t total = (size_t)Nout * C;
    for (size_t i = blockIdx.x * (size_t)blockDim.x + threadIdx.x; i < total;
         i += (size_t)gridDim.x * blockDim.x) {
        const int j = (int)(i / C), c = (int)(i % C);
        float m = -__builtin_inff();
        for (int k = 0; k < KOFF; ++k) {
            if (mask[(size_t)k * Nout + j]) {
                const float v = x[(size_t)nbr[(size_t)k * Nout + j] * C + c];
                m = fmaxf(m, v);
            }
        }
        out[i] = m;
    }
}

struct SlicePack {
    const float* outs[6];
    const int*   sl[6];
};

__global__ void slice_accum_kernel(SlicePack p, const int* __restrict__ f2v,
                                   float* __restrict__ emb, int N0) {
    const size_t total = (size_t)N0 * 528;
    for (size_t i = blockIdx.x * (size_t)blockDim.x + threadIdx.x; i < total;
         i += (size_t)gridDim.x * blockDim.x) {
        const int n = (int)(i / 528), col = (int)(i % 528);
        int L, off, ch;
        if      (col <  32) { L = 0; off =   0; ch =  32; }
        else if (col <  80) { L = 1; off =  32; ch =  48; }
        else if (col < 144) { L = 2; off =  80; ch =  64; }
        else if (col < 240) { L = 3; off = 144; ch =  96; }
        else if (col < 368) { L = 4; off = 240; ch = 128; }
        else                { L = 5; off = 368; ch = 160; }
        const int r = p.sl[L][n];
        const float v = p.outs[L][(size_t)r * ch + (col - off)];
        atomicAdd(&emb[(size_t)f2v[n] * 528 + col], v);
    }
}

__global__ void global_pool_kernel(const float* __restrict__ y,
                                   const int* __restrict__ bidx,
                                   int Nrows, int C, float* __restrict__ h) {
    const int idx = blockIdx.x * blockDim.x + threadIdx.x;
    if (idx >= 8 * C) return;
    const int bb = idx / C, c = idx % C;
    float mx = -__builtin_inff(), sum = 0.f, cnt = 0.f;
    for (int r = 0; r < Nrows; ++r) {
        if (bidx[r] == bb) {
            const float v = y[(size_t)r * C + c];
            mx = fmaxf(mx, v); sum += v; cnt += 1.f;
        }
    }
    h[(size_t)bb * (2 * C) + c]     = mx;
    h[(size_t)bb * (2 * C) + C + c] = sum / fmaxf(cnt, 1.f);
}

__global__ void fc_kernel(const float* __restrict__ in, const float* __restrict__ W,
                          const float* __restrict__ bias, float* __restrict__ out,
                          int Bn, int Cin, int Cout) {
    const int idx = blockIdx.x * blockDim.x + threadIdx.x;
    if (idx >= Bn * Cout) return;
    const int bb = idx / Cout, o = idx % Cout;
    float s = bias ? bias[o] : 0.f;
    for (int i = 0; i < Cin; ++i)
        s += in[(size_t)bb * Cin + i] * W[(size_t)i * Cout + o];
    out[idx] = s;
}

// ---------------------------------------------------------------------------
// Host-side orchestration
// ---------------------------------------------------------------------------
extern "C" void kernel_launch(void* const* d_in, const int* in_sizes, int n_in,
                              void* d_out, int out_size, void* d_ws, size_t ws_size,
                              hipStream_t stream) {
    (void)in_sizes; (void)n_in; (void)out_size; (void)ws_size;

    static const int CH[7]       = {16, 32, 48, 64, 96, 128, 160};
    static const int CONV_N[6]   = {60000, 30000, 15000, 7500, 3750, 1875};
    static const int POOL_N[6]   = {30000, 15000, 7500, 3750, 1875, 937};
    static const int EMB_N_IN[3] = {60000, 30000, 15000};
    static const int EMB_N_OUT[3]= {30000, 15000, 7500};
    static const int EMB_CH[4]   = {528, 256, 512, 1024};
    const int N0 = 100000, NV = 60000;

    // ---- unpack inputs in setup_inputs() dict order -----------------------
    int p = 0;
    const float* feats  = (const float*)d_in[p++];
    const float* mlp1_w = (const float*)d_in[p++];
    const float* mlp1_g = (const float*)d_in[p++];
    const float* mlp1_b = (const float*)d_in[p++];
    const float *convW[6], *conv_g[6], *conv_b[6];
    for (int i = 0; i < 6; ++i) convW[i]  = (const float*)d_in[p++];
    for (int i = 0; i < 6; ++i) conv_g[i] = (const float*)d_in[p++];
    for (int i = 0; i < 6; ++i) conv_b[i] = (const float*)d_in[p++];
    const float *embW[3], *emb_g[3], *emb_b[3];
    for (int i = 0; i < 3; ++i) embW[i]  = (const float*)d_in[p++];
    for (int i = 0; i < 3; ++i) emb_g[i] = (const float*)d_in[p++];
    for (int i = 0; i < 3; ++i) emb_b[i] = (const float*)d_in[p++];
    const float* f1_w = (const float*)d_in[p++];
    const float* f1_g = (const float*)d_in[p++];
    const float* f1_b = (const float*)d_in[p++];
    const float* f2_w = (const float*)d_in[p++];
    const float* f2_g = (const float*)d_in[p++];
    const float* f2_b = (const float*)d_in[p++];
    const float* f3_w = (const float*)d_in[p++];
    const float* f3_bias = (const float*)d_in[p++];
    const int* f2v = (const int*)d_in[p++];
    const int *cnbr[6], *pnbr[6], *sl6[6], *enbr[3];
    for (int i = 0; i < 6; ++i) cnbr[i] = (const int*)d_in[p++];
    for (int i = 0; i < 6; ++i) pnbr[i] = (const int*)d_in[p++];
    for (int i = 0; i < 6; ++i) sl6[i]  = (const int*)d_in[p++];
    for (int i = 0; i < 3; ++i) enbr[i] = (const int*)d_in[p++];
    const int* batch_idx = (const int*)d_in[p++];
    const unsigned char *cmask[6], *pmask[6], *emask[3];
    for (int i = 0; i < 6; ++i) cmask[i] = (const unsigned char*)d_in[p++];
    for (int i = 0; i < 6; ++i) pmask[i] = (const unsigned char*)d_in[p++];
    for (int i = 0; i < 3; ++i) emask[i] = (const unsigned char*)d_in[p++];

    // ---- workspace layout -------------------------------------------------
    char* wsb = (char*)d_ws; size_t wo = 0;
    auto walloc = [&](size_t elems) -> float* {
        float* r = (float*)(wsb + wo);
        wo += ((elems * sizeof(float) + 255) / 256) * 256;
        return r;
    };
    float* stats   = walloc(2048);
    float* xf      = walloc((size_t)N0 * 16);
    float* counts  = walloc(NV);
    float* v0      = walloc((size_t)NV * 16);
    float* convtmp = walloc((size_t)60000 * 32);     // max conv-out elems
    float* pooled[6];
    for (int i = 0; i < 6; ++i) pooled[i] = walloc((size_t)POOL_N[i] * CH[i + 1]);
    float* embin = walloc((size_t)NV * 528);
    float* ebuf[3];
    for (int i = 0; i < 3; ++i) ebuf[i] = walloc((size_t)EMB_N_OUT[i] * EMB_CH[i + 1]);
    float* h   = walloc(8 * 2048);
    float* fc1 = walloc(8 * 512);
    float* fc2 = walloc(8 * 256);

    auto gsblocks = [](size_t total) -> int {
        size_t b = (total + 255) / 256;
        return (int)(b > 4096 ? 4096 : b);
    };
    auto run_bn = [&](float* x, int N, int C, const float* g, const float* b) {
        bn_stats_kernel<<<C, 256, 0, stream>>>(x, N, C, stats);
        bn_lrelu_kernel<<<gsblocks((size_t)N * C), 256, 0, stream>>>(x, N, C, stats, g, b);
    };
    auto run_conv = [&](const float* x, const int* nbr, const unsigned char* mask,
                        const float* W, float* out, int Nin, int Nout, int Cin, int Cout) {
        dim3 grid((Nout + 31) / 32, (Cout + 127) / 128);
        sparse_conv_wmma_kernel<<<grid, 128, 0, stream>>>(x, nbr, mask, W, out,
                                                          Nin, Nout, Cin, Cout);
    };

    // ---- 1) mlp1 + BN + LReLU on the field --------------------------------
    mlp1_kernel<<<(N0 + 255) / 256, 256, 0, stream>>>(feats, mlp1_w, xf, N0);
    run_bn(xf, N0, 16, mlp1_g, mlp1_b);

    // ---- 2) field -> voxel seg_mean ---------------------------------------
    zero_kernel<<<gsblocks((size_t)NV * 16), 256, 0, stream>>>(v0, (size_t)NV * 16);
    zero_kernel<<<gsblocks(NV), 256, 0, stream>>>(counts, NV);
    seg_accum_kernel<<<gsblocks((size_t)N0 * 16), 256, 0, stream>>>(xf, f2v, N0, 16, v0, counts, 1);
    seg_div_kernel<<<gsblocks((size_t)NV * 16), 256, 0, stream>>>(v0, counts, NV, 16);

    // ---- 3) six conv+BN+LReLU+maxpool levels ------------------------------
    const float* cur = v0;
    for (int i = 0; i < 6; ++i) {
        run_conv(cur, cnbr[i], cmask[i], convW[i], convtmp,
                 CONV_N[i], CONV_N[i], CH[i], CH[i + 1]);
        run_bn(convtmp, CONV_N[i], CH[i + 1], conv_g[i], conv_b[i]);
        maxpool_kernel<<<gsblocks((size_t)POOL_N[i] * CH[i + 1]), 256, 0, stream>>>(
            convtmp, pnbr[i], pmask[i], pooled[i], POOL_N[i], CH[i + 1]);
        cur = pooled[i];
    }

    // ---- 4) slice back to field, seg_mean to voxels [NV,528] --------------
    zero_kernel<<<gsblocks((size_t)NV * 528), 256, 0, stream>>>(embin, (size_t)NV * 528);
    SlicePack sp;
    for (int i = 0; i < 6; ++i) { sp.outs[i] = pooled[i]; sp.sl[i] = sl6[i]; }
    slice_accum_kernel<<<gsblocks((size_t)N0 * 528), 256, 0, stream>>>(sp, f2v, embin, N0);
    seg_div_kernel<<<gsblocks((size_t)NV * 528), 256, 0, stream>>>(embin, counts, NV, 528);

    // ---- 5) three embedding convs (the FLOP-dominant WMMA GEMMs) ----------
    const float* ecur = embin;
    for (int i = 0; i < 3; ++i) {
        run_conv(ecur, enbr[i], emask[i], embW[i], ebuf[i],
                 EMB_N_IN[i], EMB_N_OUT[i], EMB_CH[i], EMB_CH[i + 1]);
        run_bn(ebuf[i], EMB_N_OUT[i], EMB_CH[i + 1], emb_g[i], emb_b[i]);
        ecur = ebuf[i];
    }

    // ---- 6) global max+mean pool over batch -> [8,2048] -------------------
    global_pool_kernel<<<(8 * 1024 + 255) / 256, 256, 0, stream>>>(
        ebuf[2], batch_idx, EMB_N_OUT[2], 1024, h);

    // ---- 7) FC head -------------------------------------------------------
    fc_kernel<<<(8 * 512 + 255) / 256, 256, 0, stream>>>(h, f1_w, nullptr, fc1, 8, 2048, 512);
    run_bn(fc1, 8, 512, f1_g, f1_b);
    fc_kernel<<<(8 * 256 + 255) / 256, 256, 0, stream>>>(fc1, f2_w, nullptr, fc2, 8, 512, 256);
    run_bn(fc2, 8, 256, f2_g, f2_b);
    fc_kernel<<<(8 * 20 + 255) / 256, 256, 0, stream>>>(fc2, f3_w, f3_bias, (float*)d_out, 8, 256, 20);
}
